// CausalSelfAttention_19198503813131
// MI455X (gfx1250) — matrix operand order repaired and checked by reading, not compiled, not run on hardware
//
#include <hip/hip_runtime.h>

// ---------------------------------------------------------------------------
// CDNA5 (gfx1250, wave32) bf16-WMMA causal self-attention forward.
// WMMA GEMMs + flash attention; TDM tensor_load_to_lds for K/V tiles;
// global_load_async_to_lds_b128 for GEMM tile staging.
// ---------------------------------------------------------------------------

typedef __bf16 bf16_t;
typedef __attribute__((ext_vector_type(16))) __bf16 v16bf;
typedef __attribute__((ext_vector_type(8)))  __bf16 v8bf;
typedef __attribute__((ext_vector_type(8)))  float  v8f;
typedef __attribute__((ext_vector_type(4)))  unsigned int u32x4;
typedef __attribute__((ext_vector_type(8)))  int i32x8;

#define DEVINL __device__ __forceinline__
#define AS3 __attribute__((address_space(3)))

static constexpr int Bb = 4;
static constexpr int Tt = 2048;
static constexpr int Cc = 1024;
static constexpr int Hh = 16;
static constexpr int Dd = 64;
static constexpr int BT = Bb * Tt;          // 8192

// ----- WMMA wrapper ---------------------------------------------------------
DEVINL v8f wmma_bf16(v16bf a, v16bf b, v8f c) {
  return __builtin_amdgcn_wmma_f32_16x16x32_bf16(false, a, false, b,
                                                 (short)0, c, false, false);
}

DEVINL v8f zero8() {
  v8f z;
#pragma unroll
  for (int i = 0; i < 8; ++i) z[i] = 0.0f;
  return z;
}

// ----- LDS byte-offset of a __shared__ object -------------------------------
DEVINL unsigned lds_off(const void* p) {
  return (unsigned)(unsigned long long)(AS3 const void*)p;
}

// ----- CDNA5 async copy: global -> LDS, 16 bytes per lane (ASYNCcnt) --------
DEVINL void async_b128(unsigned lds, const void* g) {
  asm volatile("global_load_async_to_lds_b128 %0, %1, off"
               :: "v"(lds), "v"((unsigned long long)(__UINTPTR_TYPE__)g)
               : "memory");
}
DEVINL void async_b128_o16(unsigned lds, const void* g) {
  // INST_OFFSET applies to both LDS and global address (ISA 08 §4.4)
  asm volatile("global_load_async_to_lds_b128 %0, %1, off offset:16"
               :: "v"(lds), "v"((unsigned long long)(__UINTPTR_TYPE__)g)
               : "memory");
}
DEVINL void wait_asynccnt0() {
#if __has_builtin(__builtin_amdgcn_s_wait_asynccnt)
  __builtin_amdgcn_s_wait_asynccnt(0);
#else
  asm volatile("s_wait_asynccnt 0x0" ::: "memory");
#endif
}
DEVINL void wait_tensorcnt0() {
#if __has_builtin(__builtin_amdgcn_s_wait_tensorcnt)
  __builtin_amdgcn_s_wait_tensorcnt(0);
#else
  asm volatile("s_wait_tensorcnt 0x0" ::: "memory");
#endif
}

// ----- TDM: DMA a 2D bf16 tile (tile_d1 rows x tile_d0 cols) into LDS -------
// D# per CDNA5 ISA, 2-group form (2D tensor, VADDR2/3 = NULL).
// LDS padding: pad_interval 32 DWORDs (value 4), pad_amount 4 DWORDs (value 3)
// => rows of 128B land with a 144B pitch, matching [rows][72] bf16 LDS arrays.
DEVINL void tdm_load_2d(unsigned ldsOff, const void* gtile,
                        unsigned tensor_d0, unsigned tensor_d1,
                        unsigned tile_d0, unsigned tile_d1, unsigned stride0) {
  unsigned long long ga = (unsigned long long)(__UINTPTR_TYPE__)gtile;
  u32x4 g0;
  g0[0] = 1u;                                            // count=1, user D#
  g0[1] = ldsOff;                                        // lds_addr
  g0[2] = (unsigned)ga;                                  // global_addr[31:0]
  g0[3] = (unsigned)((ga >> 32) & 0x01FFFFFFull) | (2u << 30); // addr[56:32]|type=2
  i32x8 g1;
  g1[0] = (int)((1u << 16) | (1u << 20) | (4u << 22) | (3u << 25));
  //             data_size=2B  pad_en     interval=32DW  amount=4DW
  g1[1] = (int)(tensor_d0 << 16);                        // tensor_dim0 @ [79:48]
  g1[2] = (int)((tensor_d0 >> 16) | (tensor_d1 << 16));  // tensor_dim1 @ [111:80]
  g1[3] = (int)((tensor_d1 >> 16) | (tile_d0 << 16));    // tile_dim0   @ [127:112]
  g1[4] = (int)(tile_d1 & 0xFFFFu);                      // tile_dim1   @ [143:128]
  g1[5] = (int)stride0;                                  // dim0 stride @ [207:160]
  g1[6] = 0;
  g1[7] = 0;
  asm volatile("tensor_load_to_lds %0, %1" :: "s"(g0), "s"(g1) : "memory");
}

// ----- Fragment loaders (row-major bf16 buffer, leading dim `ld` elems) -----
DEVINL v16bf ld_frag_a(const bf16_t* p, int ld, int row0, int k0) {
  const int lane = threadIdx.x & 31;
  const bf16_t* q = p + (size_t)(row0 + (lane & 15)) * ld + k0 + ((lane >> 4) << 3);
  v8bf lo = *(const v8bf*)q;
  v8bf hi = *(const v8bf*)(q + 16);
  return __builtin_shufflevector(lo, hi, 0,1,2,3,4,5,6,7,8,9,10,11,12,13,14,15);
}
DEVINL v16bf ld_frag_b(const bf16_t* p, int ld, int col0, int k0) {
  const int lane = threadIdx.x & 31;
  const bf16_t* q = p + (size_t)(col0 + (lane & 15)) * ld + k0 + ((lane >> 4) << 3);
  v8bf lo = *(const v8bf*)q;
  v8bf hi = *(const v8bf*)(q + 16);
  return __builtin_shufflevector(lo, hi, 0,1,2,3,4,5,6,7,8,9,10,11,12,13,14,15);
}

// ----- half-wave (16-lane) reductions ---------------------------------------
DEVINL float half_max(float v) {
#pragma unroll
  for (int m = 1; m < 16; m <<= 1) v = fmaxf(v, __shfl_xor(v, m));
  return v;
}
DEVINL float half_sum(float v) {
#pragma unroll
  for (int m = 1; m < 16; m <<= 1) v += __shfl_xor(v, m);
  return v;
}

DEVINL void cvt_store(float* p, float v)  { *p = v; }
DEVINL void cvt_store(bf16_t* p, float v) { *p = (bf16_t)v; }

// ===========================================================================
// Kernel 1: fp32 -> bf16 pack
// ===========================================================================
__global__ void cvt_f32_bf16(const float* __restrict__ s, bf16_t* __restrict__ d, int n) {
  int i = blockIdx.x * blockDim.x + threadIdx.x;
  if (i < n) d[i] = (bf16_t)s[i];
}

// ===========================================================================
// Kernel 2: C[M,N] (fp32 or bf16) = A_bf16[M,K] @ W_bf16[N,K]^T
// 256 threads = 8 waves; block tile 128x128; K-step 32; async LDS staging.
// ===========================================================================
template <typename OUT_T>
__global__ __launch_bounds__(256) void gemm_bf16_nt(const bf16_t* __restrict__ A,
                                                    const bf16_t* __restrict__ W,
                                                    OUT_T* __restrict__ C,
                                                    int M, int N, int K) {
  __shared__ bf16_t As[128][40];
  __shared__ bf16_t Bs[128][40];

  const int tid  = threadIdx.x;
  const int wave = tid >> 5;
  const int wm   = wave & 1;
  const int wn   = wave >> 1;
  const int m0   = blockIdx.y * 128;
  const int n0   = blockIdx.x * 128;

  const int lrow = tid >> 1;          // 0..127
  const int lcol = (tid & 1) * 16;    // 0 or 16

  v8f acc[4][2];
#pragma unroll
  for (int mi = 0; mi < 4; ++mi)
#pragma unroll
    for (int ni = 0; ni < 2; ++ni) acc[mi][ni] = zero8();

  for (int k0 = 0; k0 < K; k0 += 32) {
    const bf16_t* asrc = A + (size_t)(m0 + lrow) * K + k0 + lcol;
    const bf16_t* bsrc = W + (size_t)(n0 + lrow) * K + k0 + lcol;
    const unsigned la = lds_off(&As[lrow][lcol]);
    const unsigned lb = lds_off(&Bs[lrow][lcol]);
    async_b128(la, asrc);
    async_b128_o16(la, asrc);
    async_b128(lb, bsrc);
    async_b128_o16(lb, bsrc);
    if (k0 + 32 < K) {                 // pull next K-tile toward L2
      __builtin_prefetch(asrc + 32, 0, 1);
      __builtin_prefetch(bsrc + 32, 0, 1);
    }
    wait_asynccnt0();
    __syncthreads();

    v16bf fa[4], fb[2];
#pragma unroll
    for (int mi = 0; mi < 4; ++mi) fa[mi] = ld_frag_a(&As[0][0], 40, wm * 64 + mi * 16, 0);
#pragma unroll
    for (int ni = 0; ni < 2; ++ni) fb[ni] = ld_frag_b(&Bs[0][0], 40, wn * 32 + ni * 16, 0);
#pragma unroll
    for (int mi = 0; mi < 4; ++mi)
#pragma unroll
      for (int ni = 0; ni < 2; ++ni) acc[mi][ni] = wmma_bf16(fa[mi], fb[ni], acc[mi][ni]);
    __syncthreads();
  }

  const int lane = tid & 31;
  const int cn   = lane & 15;
  const int rb   = (lane >> 4) * 8;
#pragma unroll
  for (int mi = 0; mi < 4; ++mi)
#pragma unroll
    for (int ni = 0; ni < 2; ++ni)
#pragma unroll
      for (int r = 0; r < 8; ++r) {
        int row = m0 + wm * 64 + mi * 16 + rb + r;
        int col = n0 + wn * 32 + ni * 16 + cn;
        cvt_store(&C[(size_t)row * N + col], acc[mi][ni][r]);
      }
}

// ===========================================================================
// Kernel 3: per (b,t,h) row: RMS-norm over D=64, RoPE, relayout to [B,H,T,D].
// ===========================================================================
__global__ void qk_norm_rope(const bf16_t* __restrict__ in, bf16_t* __restrict__ out) {
  const int lane  = threadIdx.x & 31;
  const int w     = threadIdx.x >> 5;
  const int rowid = blockIdx.x * 8 + w;       // over B*T*H
  const int h  = rowid & 15;
  const int bt = rowid >> 4;
  const int t  = bt & (Tt - 1);
  const int b  = bt >> 11;

  const bf16_t* src = in + (size_t)bt * Cc + h * Dd;
  float x1 = (float)src[lane];
  float x2 = (float)src[lane + 32];

  float ss = x1 * x1 + x2 * x2;
#pragma unroll
  for (int m = 1; m < 32; m <<= 1) ss += __shfl_xor(ss, m);
  float inv = rsqrtf(ss * (1.0f / 64.0f) + 1.1920929e-7f);
  x1 *= inv; x2 *= inv;

  float ang = (float)t * __expf((float)lane * (-9.210340371976184f / 32.0f));
  float sn, csv;
  __sincosf(ang, &sn, &csv);
  float y1 =  x1 * csv + x2 * sn;
  float y2 = -x1 * sn  + x2 * csv;

  bf16_t* dst = out + (((size_t)b * Hh + h) * Tt + t) * Dd;
  dst[lane]      = (bf16_t)y1;
  dst[lane + 32] = (bf16_t)y2;
}

// ===========================================================================
// Kernel 4: v = (1-lamb)*v + lamb*v1, relayout to V^T = [B,H,D,T] bf16.
// ===========================================================================
__global__ void v_mix_transpose(const bf16_t* __restrict__ vin, const float* __restrict__ v1,
                                const float* __restrict__ lambp, bf16_t* __restrict__ out) {
  int idx = blockIdx.x * 256 + threadIdx.x;   // b|h|d|t bit fields
  float lam = lambp[0];
  int t = idx & (Tt - 1);
  int d = (idx >> 11) & 63;
  int h = (idx >> 17) & 15;
  int b = idx >> 21;
  size_t src = ((size_t)b * Tt + t) * Cc + h * Dd + d;
  float v = (float)vin[src];
  out[idx] = (bf16_t)((1.0f - lam) * v + lam * v1[src]);
}

// ===========================================================================
// Kernel 5: flash attention.  Grid (T/128, H, B); 8 waves x 16 query rows.
// K/V tiles delivered by TDM into padded LDS; online softmax; WMMA chains.
// ===========================================================================
__global__ __launch_bounds__(256) void flash_attn(const bf16_t* __restrict__ Q,
                                                  const bf16_t* __restrict__ Km,
                                                  const bf16_t* __restrict__ Vt,
                                                  bf16_t* __restrict__ O) {
  __shared__ bf16_t Ks[64][72];   // K rows  [s][d], 144B pitch (TDM-padded)
  __shared__ bf16_t Vs[64][72];   // V^T rows [d][s]
  __shared__ bf16_t Ps[8][16][72];

  const int b  = blockIdx.z, h = blockIdx.y;
  const int t0 = blockIdx.x * 128;
  const int tid  = threadIdx.x;
  const int w    = tid >> 5;
  const int lane = tid & 31;
  const int cn   = lane & 15;
  const int rb   = (lane >> 4) * 8;
  const int bh   = b * Hh + h;

  const bf16_t* qbase = Q + (size_t)bh * Tt * Dd;
  const int qrow0 = t0 + w * 16;
  const v16bf fq0 = ld_frag_a(qbase, Dd, qrow0, 0);
  const v16bf fq1 = ld_frag_a(qbase, Dd, qrow0, 32);

  v8f oacc[4];
#pragma unroll
  for (int j = 0; j < 4; ++j) oacc[j] = zero8();
  float m_i[8], l_i[8];
#pragma unroll
  for (int r = 0; r < 8; ++r) { m_i[r] = -1e30f; l_i[r] = 0.0f; }

  const int nblk = (t0 >> 6) + 2;              // causal: s <= t0+127
  for (int sb = 0; sb < nblk; ++sb) {
    const int s0 = sb * 64;
    if (w == 0) {   // one TDM descriptor per tile instead of 256x per-lane loads
      tdm_load_2d(lds_off(&Ks[0][0]), Km + ((size_t)bh * Tt + s0) * Dd,
                  /*tensor_d0=*/Dd, /*tensor_d1=*/Tt,
                  /*tile_d0=*/64, /*tile_d1=*/64, /*stride0=*/Dd);
      tdm_load_2d(lds_off(&Vs[0][0]), Vt + (size_t)bh * Dd * Tt + s0,
                  /*tensor_d0=*/Tt, /*tensor_d1=*/Dd,
                  /*tile_d0=*/64, /*tile_d1=*/64, /*stride0=*/Tt);
      wait_tensorcnt0();
    }
    __syncthreads();

    // ---- S = Q K^T: preload all K fragments, WMMAs back-to-back ----
    v16bf kf0[4], kf1[4];
#pragma unroll
    for (int j = 0; j < 4; ++j) {
      kf0[j] = ld_frag_b(&Ks[0][0], 72, j * 16, 0);
      kf1[j] = ld_frag_b(&Ks[0][0], 72, j * 16, 32);
    }
    v8f sacc[4];
#pragma unroll
    for (int j = 0; j < 4; ++j) {
      v8f s = zero8();
      s = wmma_bf16(fq0, kf0[j], s);
      s = wmma_bf16(fq1, kf1[j], s);
      sacc[j] = s;
    }

    // ---- preload V fragments now: LDS latency hides behind softmax ----
    v16bf vf0[4], vf1[4];
#pragma unroll
    for (int j = 0; j < 4; ++j) {
      vf0[j] = ld_frag_b(&Vs[0][0], 72, j * 16, 0);
      vf1[j] = ld_frag_b(&Vs[0][0], 72, j * 16, 32);
    }

    // ---- online softmax per row (rows live in 16-lane groups) ----
#pragma unroll
    for (int r = 0; r < 8; ++r) {
      const int trow = qrow0 + rb + r;
      float mx = -1e30f;
      float vj[4];
#pragma unroll
      for (int j = 0; j < 4; ++j) {
        int col = s0 + j * 16 + cn;
        float v = sacc[j][r] * 0.125f;       // 1/sqrt(64)
        v = (col <= trow) ? v : -1e30f;      // causal mask
        vj[j] = v;
        mx = fmaxf(mx, v);
      }
      mx = half_max(mx);
      float mnew  = fmaxf(m_i[r], mx);
      float alpha = __expf(m_i[r] - mnew);
      float ps = 0.0f;
#pragma unroll
      for (int j = 0; j < 4; ++j) {
        float pv = __expf(vj[j] - mnew);
        ps += pv;
        Ps[w][rb + r][j * 16 + cn] = (bf16_t)pv;   // wave-private relayout
      }
      ps = half_sum(ps);
      l_i[r] = l_i[r] * alpha + ps;
      m_i[r] = mnew;
#pragma unroll
      for (int j = 0; j < 4; ++j) oacc[j][r] *= alpha;
    }

    // ---- O += P V ----
    const v16bf fp0 = ld_frag_a(&Ps[w][0][0], 72, 0, 0);
    const v16bf fp1 = ld_frag_a(&Ps[w][0][0], 72, 0, 32);
#pragma unroll
    for (int j = 0; j < 4; ++j) {
      oacc[j] = wmma_bf16(fp0, vf0[j], oacc[j]);
      oacc[j] = wmma_bf16(fp1, vf1[j], oacc[j]);
    }
    __syncthreads();
  }

  bf16_t* obase = O + (size_t)b * Tt * Cc + h * Dd;
#pragma unroll
  for (int r = 0; r < 8; ++r) {
    const int trow = qrow0 + rb + r;
    const float rl = 1.0f / l_i[r];
#pragma unroll
    for (int j = 0; j < 4; ++j)
      obase[(size_t)trow * Cc + j * 16 + cn] = (bf16_t)(oacc[j][r] * rl);
  }
}

// ===========================================================================
// Host launch
// ===========================================================================
extern "C" void kernel_launch(void* const* d_in, const int* in_sizes, int n_in,
                              void* d_out, int out_size, void* d_ws, size_t ws_size,
                              hipStream_t stream) {
  (void)in_sizes; (void)n_in; (void)out_size; (void)ws_size;
  const float* x     = (const float*)d_in[0];
  const float* v1    = (const float*)d_in[1];
  const float* Wq    = (const float*)d_in[2];
  const float* Wk    = (const float*)d_in[3];
  const float* Wv    = (const float*)d_in[4];
  const float* Wproj = (const float*)d_in[5];
  const float* lamb  = (const float*)d_in[6];
  float* out = (float*)d_out;

  size_t off = 0;
  auto take = [&](size_t bytes) -> void* {
    void* p = (char*)d_ws + off;
    off += (bytes + 255) & ~(size_t)255;
    return p;
  };
  const size_t mat = (size_t)BT * Cc * sizeof(bf16_t);   // 16 MB
  const size_t wb  = (size_t)Cc * Cc * sizeof(bf16_t);   //  2 MB
  bf16_t* xb   = (bf16_t*)take(mat);
  bf16_t* wqb  = (bf16_t*)take(wb);
  bf16_t* wkb  = (bf16_t*)take(wb);
  bf16_t* wvb  = (bf16_t*)take(wb);
  bf16_t* wpb  = (bf16_t*)take(wb);
  bf16_t* qtmp = (bf16_t*)take(mat);
  bf16_t* ktmp = (bf16_t*)take(mat);
  bf16_t* vtmp = (bf16_t*)take(mat);
  bf16_t* qh   = (bf16_t*)take(mat);    // [B,H,T,D]
  bf16_t* kh   = (bf16_t*)take(mat);    // [B,H,T,D]
  bf16_t* vt   = (bf16_t*)take(mat);    // [B,H,D,T]
  bf16_t* ao   = (bf16_t*)take(mat);    // attention output, [B*T, C]

  const int nX = BT * Cc, nW = Cc * Cc;
  cvt_f32_bf16<<<nX / 256, 256, 0, stream>>>(x,     xb,  nX);
  cvt_f32_bf16<<<nW / 256, 256, 0, stream>>>(Wq,    wqb, nW);
  cvt_f32_bf16<<<nW / 256, 256, 0, stream>>>(Wk,    wkb, nW);
  cvt_f32_bf16<<<nW / 256, 256, 0, stream>>>(Wv,    wvb, nW);
  cvt_f32_bf16<<<nW / 256, 256, 0, stream>>>(Wproj, wpb, nW);

  dim3 gg(Cc / 128, BT / 128);
  gemm_bf16_nt<bf16_t><<<gg, 256, 0, stream>>>(xb, wqb, qtmp, BT, Cc, Cc);
  gemm_bf16_nt<bf16_t><<<gg, 256, 0, stream>>>(xb, wkb, ktmp, BT, Cc, Cc);
  gemm_bf16_nt<bf16_t><<<gg, 256, 0, stream>>>(xb, wvb, vtmp, BT, Cc, Cc);

  qk_norm_rope<<<(BT * Hh) / 8, 256, 0, stream>>>(qtmp, qh);
  qk_norm_rope<<<(BT * Hh) / 8, 256, 0, stream>>>(ktmp, kh);
  v_mix_transpose<<<nX / 256, 256, 0, stream>>>(vtmp, v1, lamb, vt);

  flash_attn<<<dim3(Tt / 128, Hh, Bb), 256, 0, stream>>>(qh, kh, vt, ao);

  gemm_bf16_nt<float><<<gg, 256, 0, stream>>>(ao, wpb, out, BT, Cc, Cc);

  hipMemcpyAsync(out + (size_t)BT * Cc, v1, (size_t)BT * Cc * sizeof(float),
                 hipMemcpyDeviceToDevice, stream);
}
